// NKATGaugeInvariantAttention_2121713844848
// MI455X (gfx1250) — matrix operand order repaired and checked by reading, not compiled, not run on hardware
//
#include <hip/hip_runtime.h>
#include <hip/hip_bf16.h>

typedef __attribute__((ext_vector_type(16))) _Float16 v16h;
typedef __attribute__((ext_vector_type(8)))  _Float16 v8h;
typedef __attribute__((ext_vector_type(4)))  _Float16 v4h;
typedef __attribute__((ext_vector_type(8)))  float    v8f;
typedef __attribute__((ext_vector_type(4)))  float    v4f;

#define DMODEL 1024
#define NHEAD  16
#define DK     64
#define NB     2
#define SEQ    2048
#define MR     (NB * SEQ)   // 4096 total rows

// ---------------------------------------------------------------------------
// Async global->LDS staging (CDNA5 GLOBAL_LOAD_ASYNC_TO_LDS_B128, ASYNCcnt).
// Falls back to a synchronous register copy when the builtin is unavailable.
// Builtin signature (from hipcc diagnostic): param0 = global int4*,
// so we cast to vector_size(16) int pointers with explicit address spaces.
// ---------------------------------------------------------------------------
#if defined(__has_builtin)
#if __has_builtin(__builtin_amdgcn_global_load_async_to_lds_b128)
#define ATHENA_ASYNC 1
#endif
#endif

typedef int b128_t __attribute__((vector_size(16)));
typedef __attribute__((address_space(1))) b128_t g_b128;
typedef __attribute__((address_space(3))) b128_t l_b128;

static __device__ __forceinline__ void stage16(const _Float16* g, _Float16* l) {
#ifdef ATHENA_ASYNC
  // as(3) LDS address = low 32 bits of the generic (aperture) address;
  // as(1) global address = same bits as the flat address.
  __builtin_amdgcn_global_load_async_to_lds_b128(
      (g_b128*)(unsigned long long)g,
      (l_b128*)(unsigned int)(unsigned long long)l, 0, 0);
#else
  *(v8h*)l = *(const v8h*)g;
#endif
}

static __device__ __forceinline__ void stage_wait() {
#ifdef ATHENA_ASYNC
#if __has_builtin(__builtin_amdgcn_s_wait_asynccnt)
  __builtin_amdgcn_s_wait_asynccnt(0);
#else
  asm volatile("s_wait_asynccnt 0" ::: "memory");
#endif
#endif
}

// ---------------------------------------------------------------------------
// WMMA wrapper: D = A(16x32 f16) * B(32x16 f16) + C(16x16 f32)
// ---------------------------------------------------------------------------
static __device__ __forceinline__ v8f wmma16(v16h a, v16h b, v8f c) {
  return __builtin_amdgcn_wmma_f32_16x16x32_f16(
      /*neg_a=*/false, a, /*neg_b=*/false, b,
      /*c_mod=*/(short)0, c, /*reuse_a=*/false, /*reuse_b=*/false);
}

// A-matrix fragment (16xK tile in LDS, row-major, ldm halves per row).
// ISA layout: lanes 0-15 -> M=lane, K = {kbase+0..7, kbase+16..23}
//             lanes 16-31 -> M=lane-16, K = {kbase+8..15, kbase+24..31}
static __device__ __forceinline__ v16h frag_a(const _Float16* s, int row, int ldm,
                                              int kbase, int lane) {
  int m  = lane & 15;
  int ko = kbase + ((lane & 16) ? 8 : 0);
  const _Float16* p = s + (row + m) * ldm + ko;
  v8h lo = *(const v8h*)p;
  v8h hi = *(const v8h*)(p + 16);
  v16h r;
#pragma unroll
  for (int i = 0; i < 8; ++i) { r[i] = lo[i]; r[i + 8] = hi[i]; }
  return r;
}

// B-matrix fragment from an [N][K] row-major LDS tile (lane holds one N column).
// ISA layout: lanes 0-15 -> N=lane, K = {kbase+0..7, kbase+16..23}; lanes 16-31 -> K = {kbase+8..15, kbase+24..31}
static __device__ __forceinline__ v16h frag_b(const _Float16* s, int row, int ldm,
                                              int kbase, int lane) {
  int n  = lane & 15;
  int ko = kbase + ((lane & 16) ? 8 : 0);
  const _Float16* p = s + (row + n) * ldm + ko;
  v8h lo = *(const v8h*)p;
  v8h hi = *(const v8h*)(p + 16);
  v16h r;
#pragma unroll
  for (int i = 0; i < 8; ++i) { r[i] = lo[i]; r[i + 8] = hi[i]; }
  return r;
}

// ---------------------------------------------------------------------------
// Small prep kernels
// ---------------------------------------------------------------------------
__global__ void k_gauge(const float* __restrict__ ph, const float* __restrict__ am,
                        float* __restrict__ g) {
  int h = threadIdx.x;
  if (h < NHEAD) g[h] = am[h] * cosf(ph[h]);
}

__global__ void k_f32_to_f16(const float* __restrict__ in, _Float16* __restrict__ out,
                             int n4) {
  int i = blockIdx.x * blockDim.x + threadIdx.x;
  if (i >= n4) return;
  v4f v = ((const v4f*)in)[i];
  v4h o;
#pragma unroll
  for (int j = 0; j < 4; ++j) o[j] = (_Float16)v[j];
  *(v4h*)(out + (size_t)i * 4) = o;
}

// Weff[h*64+k, c] = extra * g[h] * sum_d M[h,d,k] * W[h*64+d, c]
__global__ void k_fold(const float* __restrict__ W, const float* __restrict__ M,
                       const float* __restrict__ g, _Float16* __restrict__ Weff,
                       float extra) {
  int idx = blockIdx.x * blockDim.x + threadIdx.x;   // over 1024*1024
  int row = idx >> 10;
  int col = idx & (DMODEL - 1);
  int h = row >> 6, k = row & (DK - 1);
  const float* Mh = M + (size_t)h * DK * DK;
  const float* Wh = W + (size_t)(h * DK) * DMODEL + col;
  float acc = 0.f;
#pragma unroll 8
  for (int d = 0; d < DK; ++d) acc += Mh[d * DK + k] * Wh[(size_t)d * DMODEL];
  Weff[idx] = (_Float16)(acc * g[h] * extra);
}

// ---------------------------------------------------------------------------
// GEMM: C[M,N] = A[M,K] * B[N,K]^T   (all dims multiples of 128/32)
// 256 threads = 8 waves; block tile 128x128x32; wave tile 64x32 (4x2 WMMA).
// Double-buffered LDS with async global->LDS staging.
// ---------------------------------------------------------------------------
template <bool F32OUT>
__global__ __launch_bounds__(256) void k_gemm(const _Float16* __restrict__ A,
                                              const _Float16* __restrict__ B,
                                              float* __restrict__ Cf,
                                              _Float16* __restrict__ Ch,
                                              const float* __restrict__ bias,
                                              int M, int N, int K) {
  __shared__ __align__(16) _Float16 As[2][128 * 32];
  __shared__ __align__(16) _Float16 Bs[2][128 * 32];
  const int lane = threadIdx.x & 31;
  const int wave = threadIdx.x >> 5;
  const int m0 = blockIdx.y * 128, n0 = blockIdx.x * 128;
  const int wm = (wave & 1) * 64, wn = (wave >> 1) * 32;
  const int NK = K >> 5;

  auto stage = [&](int kk, int buf) {
#pragma unroll
    for (int i = 0; i < 2; ++i) {
      int v = threadIdx.x + i * 256;          // 512 v8h vectors per tile
      int r = v >> 2, c = (v & 3) * 8;
      stage16(A + (size_t)(m0 + r) * K + kk * 32 + c, &As[buf][r * 32 + c]);
      stage16(B + (size_t)(n0 + r) * K + kk * 32 + c, &Bs[buf][r * 32 + c]);
    }
  };

  v8f acc[4][2];
  const v8f z = {0, 0, 0, 0, 0, 0, 0, 0};
#pragma unroll
  for (int i = 0; i < 4; ++i)
#pragma unroll
    for (int j = 0; j < 2; ++j) acc[i][j] = z;

  stage(0, 0);
  stage_wait();
  __syncthreads();

  for (int kk = 0; kk < NK; ++kk) {
    const int cur = kk & 1;
    if (kk + 1 < NK) {
      if (kk + 2 < NK && threadIdx.x < 128)
        __builtin_prefetch(A + (size_t)(m0 + threadIdx.x) * K + (kk + 2) * 32, 0, 3);
      stage(kk + 1, cur ^ 1);   // async DMA of next tile overlaps the WMMAs
    }
    v16h af[4], bf[2];
#pragma unroll
    for (int i = 0; i < 4; ++i) af[i] = frag_a(As[cur], wm + 16 * i, 32, 0, lane);
#pragma unroll
    for (int j = 0; j < 2; ++j) bf[j] = frag_b(Bs[cur], wn + 16 * j, 32, 0, lane);
#pragma unroll
    for (int i = 0; i < 4; ++i)
#pragma unroll
      for (int j = 0; j < 2; ++j) acc[i][j] = wmma16(af[i], bf[j], acc[i][j]);
    stage_wait();
    __syncthreads();
  }

  const int g8 = (lane & 16) ? 8 : 0;
  const int nl = lane & 15;
#pragma unroll
  for (int i = 0; i < 4; ++i)
#pragma unroll
    for (int j = 0; j < 2; ++j)
#pragma unroll
      for (int e = 0; e < 8; ++e) {
        int row = m0 + wm + 16 * i + e + g8;
        int col = n0 + wn + 16 * j + nl;
        float v = acc[i][j][e];
        if (F32OUT) Cf[(size_t)row * N + col] = v + bias[col];
        else        Ch[(size_t)row * N + col] = (_Float16)v;
      }
}

// ---------------------------------------------------------------------------
// Attention: block = 128 thr (4 waves), owns 64 query rows of one (b,h).
// PASS 1: online-softmax statistics of QK^T -> rowmax + 1/rowsum (no attn
//         traffic; scores recomputed via WMMA, 1 exp per element).
// PASS 2: p = exp(s-m)/rowsum -> NORMALIZED attn written once (the only HBM
//         pass over the 537MB attn output); ctx = p @ V via WMMA with
//         LDS-transposed V.  K/V tiles double-buffered, K staged async.
// ---------------------------------------------------------------------------
template <int PASS>
__global__ __launch_bounds__(128) void k_attn(
    const _Float16* __restrict__ Qn, const _Float16* __restrict__ Kn,
    const _Float16* __restrict__ Vp, float* __restrict__ attn,
    float* __restrict__ rowmax, float* __restrict__ invsum,
    _Float16* __restrict__ ctx) {
  __shared__ __align__(16) _Float16 Qs[64 * 64];
  __shared__ __align__(16) _Float16 Ks[2][32 * 64];
  __shared__ __align__(16) _Float16 Vt[2][64 * 40];   // [dk][n], padded ldm=40
  __shared__ __align__(16) _Float16 Ps[4][16 * 32];

  const int lane = threadIdx.x & 31;
  const int wave = threadIdx.x >> 5;
  const int m0 = blockIdx.x * 64;
  const int h = blockIdx.y;
  const int b = blockIdx.z;
  const int g8 = (lane & 16) ? 8 : 0;
  const int nl = lane & 15;

  const size_t qoff = ((size_t)b * SEQ + m0) * DMODEL + h * DK;
  const size_t kvoff = (size_t)b * SEQ * DMODEL + h * DK;

  auto stage_kv = [&](int it, int buf) {
    int nn = it * 32;
#pragma unroll
    for (int i = 0; i < 2; ++i) {
      int v = threadIdx.x + i * 128;
      int r = v >> 3, c = (v & 7) * 8;
      stage16(Kn + kvoff + (size_t)(nn + r) * DMODEL + c, &Ks[buf][r * 64 + c]);
      if (PASS == 2) {
        v8h x = *(const v8h*)(Vp + kvoff + (size_t)(nn + r) * DMODEL + c);
#pragma unroll
        for (int j = 0; j < 8; ++j) Vt[buf][(c + j) * 40 + r] = x[j];
      }
    }
  };

  // Stage Q tile (async) + first K/V tile, then build the Q fragments.
#pragma unroll
  for (int i = 0; i < 4; ++i) {
    int v = threadIdx.x + i * 128;
    int r = v >> 3, c = (v & 7) * 8;
    stage16(Qn + qoff + (size_t)r * DMODEL + c, &Qs[r * 64 + c]);
  }
  stage_kv(0, 0);
  stage_wait();
  __syncthreads();
  const v16h qf0 = frag_a(Qs, wave * 16, 64, 0, lane);
  const v16h qf1 = frag_a(Qs, wave * 16, 64, 32, lane);

  const int rowhead = (b * NHEAD + h) * SEQ;
  float mrow[8], inv[8];
  float mx8[8], sum8[8];
  v8f cacc[4];
  const v8f z = {0, 0, 0, 0, 0, 0, 0, 0};
  if (PASS == 1) {
#pragma unroll
    for (int e = 0; e < 8; ++e) { mx8[e] = -3.0e38f; sum8[e] = 0.f; }
  } else {
#pragma unroll
    for (int e = 0; e < 8; ++e) {
      mrow[e] = rowmax[rowhead + m0 + wave * 16 + g8 + e];
      inv[e]  = invsum[rowhead + m0 + wave * 16 + g8 + e];
    }
#pragma unroll
    for (int d = 0; d < 4; ++d) cacc[d] = z;
  }

  float* attn_h = attn + (size_t)(b * NHEAD + h) * SEQ * SEQ;
  const int NT = SEQ / 32;

  for (int it = 0; it < NT; ++it) {
    const int cur = it & 1;
    const int n0 = it * 32;
    if (it + 1 < NT) {
      if (it + 2 < NT)
        __builtin_prefetch(Kn + kvoff + (size_t)(n0 + 64 + (threadIdx.x & 31)) * DMODEL, 0, 3);
      stage_kv(it + 1, cur ^ 1);   // overlap next-tile DMA with this tile's WMMAs
    }

#pragma unroll
    for (int nt = 0; nt < 32; nt += 16) {
      v8f s = z;
      s = wmma16(qf0, frag_b(Ks[cur], nt, 64, 0, lane), s);
      s = wmma16(qf1, frag_b(Ks[cur], nt, 64, 32, lane), s);
      if (PASS == 1) {
        // Online (max, rescaled-sum): single exp per element.
#pragma unroll
        for (int e = 0; e < 8; ++e) {
          float sv = s[e];
          float mo = mx8[e];
          float mn = fmaxf(mo, sv);
          float ee = __expf(fminf(mo, sv) - mn);
          sum8[e] = (sv > mo) ? (sum8[e] * ee + 1.f) : (sum8[e] + ee);
          mx8[e] = mn;
        }
      } else {
#pragma unroll
        for (int e = 0; e < 8; ++e) {
          float p = __expf(s[e] - mrow[e]) * inv[e];   // fully normalized
          attn_h[(size_t)(m0 + wave * 16 + g8 + e) * SEQ + n0 + nt + nl] = p;
          Ps[wave][(g8 + e) * 32 + nt + nl] = (_Float16)p;
        }
      }
    }
    if (PASS == 2) {
      const v16h pf = frag_a(Ps[wave], 0, 32, 0, lane);
#pragma unroll
      for (int d = 0; d < 4; ++d)
        cacc[d] = wmma16(pf, frag_b(Vt[cur], d * 16, 40, 0, lane), cacc[d]);
    }
    stage_wait();
    __syncthreads();
  }

  if (PASS == 1) {
    // Merge (max, sum) pairs across the 16 lanes holding one row.
#pragma unroll
    for (int off = 8; off >= 1; off >>= 1)
#pragma unroll
      for (int e = 0; e < 8; ++e) {
        float mo2 = __shfl_xor(mx8[e], off, 32);
        float so2 = __shfl_xor(sum8[e], off, 32);
        float mn = fmaxf(mx8[e], mo2);
        sum8[e] = sum8[e] * __expf(mx8[e] - mn) + so2 * __expf(mo2 - mn);
        mx8[e] = mn;
      }
    if (nl == 0) {
#pragma unroll
      for (int e = 0; e < 8; ++e) {
        rowmax[rowhead + m0 + wave * 16 + g8 + e] = mx8[e];
        invsum[rowhead + m0 + wave * 16 + g8 + e] = 1.0f / sum8[e];
      }
    }
  } else {
    // ctx rows already normalized (p was normalized) — store directly.
#pragma unroll
    for (int d = 0; d < 4; ++d)
#pragma unroll
      for (int e = 0; e < 8; ++e)
        ctx[((size_t)b * SEQ + m0 + wave * 16 + g8 + e) * DMODEL + h * DK +
            d * 16 + nl] = (_Float16)cacc[d][e];
  }
}

// ---------------------------------------------------------------------------
extern "C" void kernel_launch(void* const* d_in, const int* in_sizes, int n_in,
                              void* d_out, int out_size, void* d_ws, size_t ws_size,
                              hipStream_t stream) {
  (void)in_sizes; (void)n_in; (void)out_size; (void)ws_size;
  const float* x  = (const float*)d_in[0];
  const float* Wq = (const float*)d_in[1];
  const float* Wk = (const float*)d_in[2];
  const float* Wv = (const float*)d_in[3];
  const float* Wo = (const float*)d_in[4];
  const float* bo = (const float*)d_in[5];
  const float* gp = (const float*)d_in[6];
  const float* ga = (const float*)d_in[7];
  const float* nc = (const float*)d_in[8];

  float* out  = (float*)d_out;
  float* attn = out + (size_t)NB * SEQ * DMODEL;

  char* w = (char*)d_ws;
  _Float16* x16  = (_Float16*)w; w += (size_t)MR * DMODEL * 2;
  _Float16* qw16 = (_Float16*)w; w += (size_t)DMODEL * DMODEL * 2;
  _Float16* kw16 = (_Float16*)w; w += (size_t)DMODEL * DMODEL * 2;
  _Float16* vw16 = (_Float16*)w; w += (size_t)DMODEL * DMODEL * 2;
  _Float16* ow16 = (_Float16*)w; w += (size_t)DMODEL * DMODEL * 2;
  _Float16* qn16 = (_Float16*)w; w += (size_t)MR * DMODEL * 2;
  _Float16* kn16 = (_Float16*)w; w += (size_t)MR * DMODEL * 2;
  _Float16* vp16 = (_Float16*)w; w += (size_t)MR * DMODEL * 2;
  _Float16* cx16 = (_Float16*)w; w += (size_t)MR * DMODEL * 2;
  float* rowmax  = (float*)w;    w += (size_t)NB * NHEAD * SEQ * 4;
  float* invsum  = (float*)w;    w += (size_t)NB * NHEAD * SEQ * 4;
  float* g       = (float*)w;    w += 256;

  k_gauge<<<1, NHEAD, 0, stream>>>(gp, ga, g);
  k_f32_to_f16<<<MR * DMODEL / 4 / 256, 256, 0, stream>>>(x, x16, MR * DMODEL / 4);
  k_f32_to_f16<<<DMODEL * DMODEL / 4 / 256, 256, 0, stream>>>(Wv, vw16, DMODEL * DMODEL / 4);
  k_f32_to_f16<<<DMODEL * DMODEL / 4 / 256, 256, 0, stream>>>(Wo, ow16, DMODEL * DMODEL / 4);
  // Fold gauge + nc_mixing (+ 1/sqrt(dk) on the Q side) into projection weights.
  k_fold<<<DMODEL * DMODEL / 256, 256, 0, stream>>>(Wq, nc, g, qw16, 0.125f);
  k_fold<<<DMODEL * DMODEL / 256, 256, 0, stream>>>(Wk, nc, g, kw16, 1.0f);

  dim3 gg(DMODEL / 128, MR / 128);
  k_gemm<false><<<gg, 256, 0, stream>>>(x16, qw16, nullptr, qn16, nullptr, MR, DMODEL, DMODEL);
  k_gemm<false><<<gg, 256, 0, stream>>>(x16, kw16, nullptr, kn16, nullptr, MR, DMODEL, DMODEL);
  k_gemm<false><<<gg, 256, 0, stream>>>(x16, vw16, nullptr, vp16, nullptr, MR, DMODEL, DMODEL);

  dim3 ag(SEQ / 64, NHEAD, NB);
  k_attn<1><<<ag, 128, 0, stream>>>(qn16, kn16, vp16, attn, rowmax, invsum, cx16);
  k_attn<2><<<ag, 128, 0, stream>>>(qn16, kn16, vp16, attn, rowmax, invsum, cx16);

  k_gemm<true><<<gg, 256, 0, stream>>>(cx16, ow16, out, nullptr, bo, MR, DMODEL, DMODEL);
}
